// HybridMemoryMultiFocalPercent_50706383896900
// MI455X (gfx1250) — matrix operand order repaired and checked by reading, compiled, not run on hardware
//
#include <hip/hip_runtime.h>
#include <hip/hip_fp16.h>

typedef __attribute__((ext_vector_type(16))) _Float16 v16h;
typedef __attribute__((ext_vector_type(8)))  _Float16 v8h;
typedef __attribute__((ext_vector_type(8)))  float    v8f;
typedef __attribute__((ext_vector_type(4)))  int      v4i;

#define NB 4096
#define NM 4096
#define ND 512
#define NC 80
#define TEMP_INV (1.0f/3.0f)
#define ALPHA 0.1f
#define NITER 100
#define TOPP 0.1f

#define BT   128   // block tile (rows and cols) for the big GEMMs
#define LDSP 40    // padded LDS row stride in halves (32 data + 8 pad)

#define WAIT_ASYNC() asm volatile("s_wait_asynccnt 0" ::: "memory")

// ---- async global->LDS copy of 16 bytes (GLOBAL_LOAD_ASYNC_TO_LDS_B128) ----
// Probe-discovered prototype: both pointer params are generic `int4*`.
__device__ __forceinline__ void async_ld16(const _Float16* g, _Float16* l) {
  __builtin_amdgcn_global_load_async_to_lds_b128((v4i*)g, (v4i*)l, 0, 0);
}

// ---------------- WMMA fragment loaders (v_wmma_f32_16x16x32_f16 layouts) ----
// A: 16x32 f16. lane r=lane&15 is row, g=lane>>4: halves 0..7 -> K=8g+0..7,
//    halves 8..15 -> K=16+8g+0..7.
__device__ __forceinline__ v16h load_frag_a_g(const _Float16* __restrict__ base, int ld,
                                              int row0, int k0, int lane) {
  int r = lane & 15, g = lane >> 4;
  const _Float16* p = base + (size_t)(row0 + r) * (size_t)ld + (size_t)(k0 + 8 * g);
  union { v16h v; v8h h[2]; } u;
  u.h[0] = *(const v8h*)(p);
  u.h[1] = *(const v8h*)(p + 16);
  return u.v;
}

typedef __attribute__((address_space(3))) const _Float16* ldsc_h;
typedef __attribute__((address_space(3))) const v8h*      ldsc_v8h;

__device__ __forceinline__ v16h lds_frag_a(const _Float16* buf, int row0, int lane) {
  int r = lane & 15, g = lane >> 4;
  ldsc_h p = (ldsc_h)(unsigned int)(unsigned long long)(buf + (row0 + r) * LDSP + 8 * g);
  union { v16h v; v8h h[2]; } u;
  u.h[0] = *(ldsc_v8h)(p);
  u.h[1] = *(ldsc_v8h)(p + 16);
  return u.v;
}
// B: 32x16 f16, operand stored transposed (row n = column n, contiguous K).
__device__ __forceinline__ v16h lds_frag_b(const _Float16* buf, int n0, int lane) {
  int n = lane & 15, g = lane >> 4;
  ldsc_h p = (ldsc_h)(unsigned int)(unsigned long long)(buf + (n0 + n) * LDSP + 8 * g);
  union { v16h v; v8h h[2]; } u;
  u.h[0] = *(ldsc_v8h)(p);
  u.h[1] = *(ldsc_v8h)(p + 16);
  return u.v;
}
// C/D: VGPR vg, lane group g: row = row0 + vg + 8g, col = n0 + (lane&15)

// ---------------- setup kernels ------------------------------------------
__global__ void k_init(const int* __restrict__ labels, const int* __restrict__ indexes,
                       int* __restrict__ tgt, int* __restrict__ nums, float* __restrict__ out) {
  int i = blockIdx.x * blockDim.x + threadIdx.x;
  if (i < NB) tgt[i] = labels[indexes[i]];
  if (i < NC) nums[i] = 0;
  if (i == 0) out[0] = 0.0f;
}

__global__ void k_pinit(const int* __restrict__ tgt, float* __restrict__ p,
                        _Float16* __restrict__ pT0) {
  int idx = blockIdx.x * blockDim.x + threadIdx.x;  // NB*NC
  if (idx >= NB * NC) return;
  int i = idx / NC, c = idx - i * NC;
  float v = (tgt[i] == c) ? 1.0f : 0.0f;            // oh * mask (t==-1 -> 0)
  p[idx] = v;
  pT0[(size_t)c * NB + i] = (_Float16)v;
}

__global__ void k_norm_results(const float* __restrict__ res, _Float16* __restrict__ rn) {
  __shared__ float red[256];
  int b = blockIdx.x, tid = threadIdx.x;
  float ss = 0.0f;
  for (int d = tid; d < ND; d += 256) { float v = res[(size_t)b * ND + d]; ss += v * v; }
  red[tid] = ss; __syncthreads();
  for (int s = 128; s > 0; s >>= 1) { if (tid < s) red[tid] += red[tid + s]; __syncthreads(); }
  float inv = 1.0f / fmaxf(sqrtf(red[0]), 1e-12f);
  for (int d = tid; d < ND; d += 256)
    rn[(size_t)b * ND + d] = (_Float16)(res[(size_t)b * ND + d] * inv);
}

__global__ void k_f2h(const float* __restrict__ f, _Float16* __restrict__ h, int n) {
  int i = blockIdx.x * blockDim.x + threadIdx.x;
  if (i < n) h[i] = (_Float16)f[i];
}

// ---------------- LDS double-buffered GEMM: C = scale*rowscale[i]*(A @ Bt^T) ----
// A: [4096][K] f16 row-major, Bt: [4096][K] f16 row-major. Block tile 128x128,
// wave tile 32x64 (2x4 accumulators). Tiles staged with async global->LDS.
__global__ void __launch_bounds__(256)
k_gemm_nt(const _Float16* __restrict__ A, const _Float16* __restrict__ Bt, int K,
          float scale, const float* __restrict__ rowscale,
          float* __restrict__ Cf, _Float16* __restrict__ Ch) {
  __shared__ _Float16 sA[2][BT * LDSP];
  __shared__ _Float16 sB[2][BT * LDSP];
  int tid = threadIdx.x, lane = tid & 31, w = tid >> 5;
  int wr = w & 3, wc = w >> 2;                   // 4 row-groups x 2 col-groups
  int gm0 = (blockIdx.x >> 5) * BT, gn0 = (blockIdx.x & 31) * BT;

  v8f acc[2][4];
#pragma unroll
  for (int i = 0; i < 2; ++i)
#pragma unroll
    for (int j = 0; j < 4; ++j) acc[i][j] = (v8f){};

  int row0c = tid >> 2, row1c = row0c + 64, sub = tid & 3;  // 16B chunks
  int nk = K >> 5;

  auto stage = [&](int kb, int buf) {
    int k0 = kb << 5;
    async_ld16(A  + (size_t)(gm0 + row0c) * K + k0 + sub * 8, &sA[buf][row0c * LDSP + sub * 8]);
    async_ld16(A  + (size_t)(gm0 + row1c) * K + k0 + sub * 8, &sA[buf][row1c * LDSP + sub * 8]);
    async_ld16(Bt + (size_t)(gn0 + row0c) * K + k0 + sub * 8, &sB[buf][row0c * LDSP + sub * 8]);
    async_ld16(Bt + (size_t)(gn0 + row1c) * K + k0 + sub * 8, &sB[buf][row1c * LDSP + sub * 8]);
  };

  stage(0, 0);
  WAIT_ASYNC(); __syncthreads();
#pragma unroll 1
  for (int kb = 0; kb < nk; ++kb) {
    int cur = kb & 1;
    if (kb + 1 < nk) stage(kb + 1, cur ^ 1);
    v16h fa[2], fb[4];
#pragma unroll
    for (int i = 0; i < 2; ++i) fa[i] = lds_frag_a(sA[cur], wr * 32 + i * 16, lane);
#pragma unroll
    for (int j = 0; j < 4; ++j) fb[j] = lds_frag_b(sB[cur], wc * 64 + j * 16, lane);
#pragma unroll
    for (int i = 0; i < 2; ++i)
#pragma unroll
      for (int j = 0; j < 4; ++j)
        acc[i][j] = __builtin_amdgcn_wmma_f32_16x16x32_f16(false, fa[i], false, fb[j],
                                                           (short)0, acc[i][j], false, false);
    WAIT_ASYNC(); __syncthreads();
  }

  int n = lane & 15, g = lane >> 4;
#pragma unroll
  for (int i = 0; i < 2; ++i)
#pragma unroll
    for (int j = 0; j < 4; ++j)
#pragma unroll
      for (int vg = 0; vg < 8; ++vg) {
        int row = gm0 + wr * 32 + i * 16 + vg + 8 * g;
        int col = gn0 + wc * 64 + j * 16 + n;
        float v = acc[i][j][vg] * scale;
        if (rowscale) v *= rowscale[row];
        if (Cf) Cf[(size_t)row * NM + col] = v;
        if (Ch) Ch[(size_t)row * NM + col] = (_Float16)v;
      }
}

__global__ void k_rownorm(const float* __restrict__ logits, float* __restrict__ ninv) {
  __shared__ float red[256];
  int i = blockIdx.x, tid = threadIdx.x;
  float ss = 0.0f;
  for (int m = tid; m < NM; m += 256) { float v = logits[(size_t)i * NM + m]; ss += v * v; }
  red[tid] = ss; __syncthreads();
  for (int s = 128; s > 0; s >>= 1) { if (tid < s) red[tid] += red[tid + s]; __syncthreads(); }
  if (tid == 0) ninv[i] = 1.0f / fmaxf(sqrtf(red[0]), 1e-30f);
}

// ---------------- propagation: p' = 0.9 p + 0.1 S@p (fused pT snapshot) ----
// 64 blocks x 8 waves. Block owns 64 rows x 80 cols; wave = 16-row stripe,
// col-tiles split 3/2 between wave pairs (wave-uniform -> EXEC all ones).
// pT tile (80 x 32) staged via async loads, shared by all waves; S streamed.
__global__ void __launch_bounds__(256)
k_prop(const _Float16* __restrict__ S, const _Float16* __restrict__ pT,
       float* __restrict__ p, _Float16* __restrict__ pTn, const int* __restrict__ tgt) {
  __shared__ _Float16 sB[2][NC * LDSP];
  int tid = threadIdx.x, lane = tid & 31, w = tid >> 5;
  int stripe = w >> 1, half = w & 1;
  int rbase = blockIdx.x * 64 + stripe * 16;
  int tstart = half ? 3 : 0;
  int nt = half ? 2 : 3;
  v8f acc[3];
#pragma unroll
  for (int j = 0; j < 3; ++j) acc[j] = (v8f){};

  int crow = tid >> 2, sub = tid & 3;   // 320 chunks: tid (<256) and 256+tid (tid<64)
  auto stage = [&](int kb, int buf) {
    int k0 = kb << 5;
    async_ld16(pT + (size_t)crow * NB + k0 + sub * 8, &sB[buf][crow * LDSP + sub * 8]);
    if (tid < 64) {
      int r2 = crow + 64;
      async_ld16(pT + (size_t)r2 * NB + k0 + sub * 8, &sB[buf][r2 * LDSP + sub * 8]);
    }
  };

  stage(0, 0);
  WAIT_ASYNC(); __syncthreads();
#pragma unroll 1
  for (int kb = 0; kb < (NB >> 5); ++kb) {
    int cur = kb & 1;
    if (kb + 1 < (NB >> 5)) stage(kb + 1, cur ^ 1);
    v16h fa = load_frag_a_g(S, NB, rbase, kb << 5, lane);
#pragma unroll
    for (int j = 0; j < 3; ++j)
      if (j < nt) {
        v16h fb = lds_frag_b(sB[cur], (tstart + j) * 16, lane);
        acc[j] = __builtin_amdgcn_wmma_f32_16x16x32_f16(false, fa, false, fb,
                                                        (short)0, acc[j], false, false);
      }
    WAIT_ASYNC(); __syncthreads();
  }

  int n = lane & 15, g = lane >> 4;
#pragma unroll
  for (int j = 0; j < 3; ++j)
    if (j < nt)
#pragma unroll
      for (int vg = 0; vg < 8; ++vg) {
        int row = rbase + vg + 8 * g;
        int col = (tstart + j) * 16 + n;
        int t = tgt[row];
        size_t o = (size_t)row * NC + col;
        float np = (t >= 0) ? (1.0f - ALPHA) * p[o] + ALPHA * acc[j][vg] : 0.0f;
        p[o] = np;
        pTn[(size_t)col * NB + row] = (_Float16)np;
      }
}

// ---------------- argmax + class counts -----------------------------------
__global__ void k_argmax(const float* __restrict__ p, int* __restrict__ prop,
                         int* __restrict__ nums) {
  int i = blockIdx.x * blockDim.x + threadIdx.x;
  if (i >= NB) return;
  float best = p[(size_t)i * NC]; int bc = 0;
  for (int c = 1; c < NC; ++c) {
    float v = p[(size_t)i * NC + c];
    if (v > best) { best = v; bc = c; }
  }
  prop[i] = bc;
  __hip_atomic_fetch_add(&nums[bc], 1, __ATOMIC_RELAXED, __HIP_MEMORY_SCOPE_AGENT);
}

// ---------------- vec[b,c] = mean over {m: prop[m]==c} of logits[b,m] ------
__global__ void k_vec(const float* __restrict__ logits, const int* __restrict__ prop,
                      const int* __restrict__ nums, float* __restrict__ vec) {
  __shared__ float acc[NC];
  int b = blockIdx.x, tid = threadIdx.x;
  if (tid < NC) acc[tid] = 0.0f;
  __syncthreads();
  for (int m = tid; m < NM; m += 256) {
    float v = logits[(size_t)b * NM + m];
    __hip_atomic_fetch_add(&acc[prop[m]], v, __ATOMIC_RELAXED, __HIP_MEMORY_SCOPE_WORKGROUP);
  }
  __syncthreads();
  if (tid < NC) {
    int nn = nums[tid];
    vec[(size_t)b * NC + tid] = acc[tid] / (float)(nn > 0 ? nn : 1);
  }
}

// ---------------- masked top-percent focal softmax + NLL -------------------
__global__ void k_loss(const float* __restrict__ vec, const int* __restrict__ nums,
                       const int* __restrict__ prop, const int* __restrict__ tgt,
                       float* __restrict__ out) {
  int b = blockIdx.x * blockDim.x + threadIdx.x;
  if (b >= NB) return;
  float neg[NC], nn[NC];
  int pb = prop[b];
  float negsum = 0.0f, pos = 0.0f;
  for (int c = 0; c < NC; ++c) {
    float e = expf(vec[(size_t)b * NC + c]);
    float me = (nums[c] > 0) ? e : 0.0f;
    if (c == pb) { pos = me; neg[c] = 0.0f; }
    else         { neg[c] = me; negsum += me; }
  }
  float ns = (negsum > 0.0f) ? negsum : 1.0f;
  for (int c = 0; c < NC; ++c) nn[c] = neg[c] / ns;
  for (int i2 = 1; i2 < NC; ++i2) {               // descending insertion sort
    float key = nn[i2]; int j = i2 - 1;
    while (j >= 0 && nn[j] < key) { nn[j + 1] = nn[j]; --j; }
    nn[j + 1] = key;
  }
  float cum = 0.0f, bestd = 3.4e38f; int kidx = 0;
  for (int c = 0; c < NC; ++c) {
    cum += nn[c];
    float d = fabsf(cum - TOPP);
    if (d < bestd) { bestd = d; kidx = c; }       // first argmin
  }
  float minv = nn[kidx] * negsum;
  float denom = 1e-6f, newt = 0.0f;
  int t = tgt[b];
  for (int c = 0; c < NC; ++c) {
    float v = (c == pb) ? pos : ((neg[c] < minv) ? 0.0f : neg[c]);
    denom += v;
    if (c == t) newt = v;
  }
  float lp = logf(newt / denom + 1e-6f);
  __hip_atomic_fetch_add(out, -lp / (float)NB, __ATOMIC_RELAXED, __HIP_MEMORY_SCOPE_AGENT);
}

// ---------------- host launcher -------------------------------------------
extern "C" void kernel_launch(void* const* d_in, const int* in_sizes, int n_in,
                              void* d_out, int out_size, void* d_ws, size_t ws_size,
                              hipStream_t stream) {
  (void)in_sizes; (void)n_in; (void)out_size; (void)ws_size;
  const float* results  = (const float*)d_in[0];
  const float* features = (const float*)d_in[1];
  const int*   labels   = (const int*)d_in[2];
  const int*   indexes  = (const int*)d_in[3];
  float* out = (float*)d_out;

  char* ws = (char*)d_ws;
  size_t off = 0;
  auto alloc = [&](size_t bytes) -> void* {
    void* pp = ws + off;
    off += (bytes + 255) & ~(size_t)255;
    return pp;
  };
  float*    logits   = (float*)   alloc((size_t)NB * NM * 4);
  _Float16* logits_h = (_Float16*)alloc((size_t)NB * NM * 2);
  _Float16* S_h      = (_Float16*)alloc((size_t)NB * NB * 2);
  _Float16* rn_h     = (_Float16*)alloc((size_t)NB * ND * 2);
  _Float16* ft_h     = (_Float16*)alloc((size_t)NM * ND * 2);
  float*    p        = (float*)   alloc((size_t)NB * NC * 4);
  _Float16* pTa      = (_Float16*)alloc((size_t)NC * NB * 2);
  _Float16* pTb      = (_Float16*)alloc((size_t)NC * NB * 2);
  float*    vec      = (float*)   alloc((size_t)NB * NC * 4);
  float*    ninv     = (float*)   alloc((size_t)NB * 4);
  int*      tgt      = (int*)     alloc((size_t)NB * 4);
  int*      prop     = (int*)     alloc((size_t)NB * 4);
  int*      nums     = (int*)     alloc((size_t)NC * 4);

  k_init<<<16, 256, 0, stream>>>(labels, indexes, tgt, nums, out);
  k_norm_results<<<NB, 256, 0, stream>>>(results, rn_h);
  k_f2h<<<(NM * ND + 255) / 256, 256, 0, stream>>>(features, ft_h, NM * ND);
  k_gemm_nt<<<1024, 256, 0, stream>>>(rn_h, ft_h, ND, TEMP_INV, nullptr, logits, logits_h);
  k_rownorm<<<NB, 256, 0, stream>>>(logits, ninv);
  k_gemm_nt<<<1024, 256, 0, stream>>>(logits_h, logits_h, NM, 1.0f, ninv, nullptr, S_h);
  k_pinit<<<(NB * NC + 255) / 256, 256, 0, stream>>>(tgt, p, pTa);
  for (int it = 0; it < NITER; ++it) {
    _Float16* src = (it & 1) ? pTb : pTa;
    _Float16* dst = (it & 1) ? pTa : pTb;
    k_prop<<<64, 256, 0, stream>>>(S_h, src, p, dst, tgt);
  }
  k_argmax<<<16, 256, 0, stream>>>(p, prop, nums);
  k_vec<<<NB, 256, 0, stream>>>(logits, prop, nums, vec);
  k_loss<<<16, 256, 0, stream>>>(vec, nums, prop, tgt, out);
}
